// CAAN_19550691131467
// MI455X (gfx1250) — matrix operand order repaired and checked by reading, not compiled, not run on hardware
//
#include <hip/hip_runtime.h>
#include <hip/hip_bf16.h>

// Problem constants (match reference: B=32, N=2048, D=256)
#define BB 32
#define NN 2048
#define DD 256

typedef __bf16 bf16_t;
typedef __attribute__((ext_vector_type(16))) __bf16 v16bf;
typedef __attribute__((ext_vector_type(8)))  float  v8f;

// ---------------------------------------------------------------------------
// WMMA fragment helpers (wave32; contiguous k = 16*half + i in both A and B)
// ---------------------------------------------------------------------------

// A matrix 16x32 bf16: lane holds row M=lane&15; half=lane>>4.
// element i -> K = half*16 + i (contiguous, matches B)
__device__ inline v16bf load_a_frag(const bf16_t* base, int ld, int lane) {
    const int m = lane & 15, half = lane >> 4;
    const bf16_t* p = base + (size_t)m * ld + half * 16;
    v16bf a;
#pragma unroll
    for (int i = 0; i < 16; ++i) a[i] = p[i];
    return a;
}

// B matrix 32x16 bf16, source stored "col-major" (base[col*ld + k]):
// lane holds col N=lane&15; element e -> K = (lane>>4)*16 + e  (contiguous!)
__device__ inline v16bf load_b_frag(const bf16_t* base, int ld, int lane) {
    const int n = lane & 15, half = lane >> 4;
    const bf16_t* p = base + (size_t)n * ld + half * 16;
    v16bf b;
#pragma unroll
    for (int i = 0; i < 16; ++i) b[i] = p[i];
    return b;
}

__device__ inline v8f wmma_bf16(v16bf a, v16bf b, v8f c) {
    // (neg_a, A, neg_b, B, c_mod, C, reuse_a, reuse_b)
    return __builtin_amdgcn_wmma_f32_16x16x32_bf16(false, a, false, b,
                                                   (short)0, c, false, false);
}

// ---------------------------------------------------------------------------
// Conversion kernels
// ---------------------------------------------------------------------------
__global__ void cvt_bf16_kernel(const float* __restrict__ src,
                                bf16_t* __restrict__ dst, int n) {
    int i = blockIdx.x * blockDim.x + threadIdx.x;
    if (i < n) dst[i] = (bf16_t)src[i];
}

// W [D,D] f32 row-major -> W^T bf16 (out[e*D+d] = in[d*D+e])
__global__ void cvt_wT_kernel(const float* __restrict__ w,
                              bf16_t* __restrict__ wt) {
    int i = blockIdx.x * blockDim.x + threadIdx.x;  // output index e*DD+d
    int e = i >> 8, d = i & (DD - 1);
    wt[i] = (bf16_t)w[d * DD + e];
}

// ---------------------------------------------------------------------------
// QKV projection: C[16x16] = X[16xD] * W[Dx16] + bias.
// One wave per output tile. mode 0: store row-major; mode 1: store transposed
// per batch ([B][D][N]) so attention B-fragments of V are contiguous.
// ---------------------------------------------------------------------------
__global__ __launch_bounds__(32)
void qkv_proj_kernel(const bf16_t* __restrict__ xb, const bf16_t* __restrict__ wT,
                     const float* __restrict__ bias, bf16_t* __restrict__ out_rm,
                     bf16_t* __restrict__ out_tr, int transpose_out) {
    const int lane = threadIdx.x;
    const int row0 = blockIdx.x * 16;   // over B*N
    const int col0 = blockIdx.y * 16;   // over D

    v8f c = {};
#pragma unroll
    for (int dk = 0; dk < DD; dk += 32) {
        v16bf a = load_a_frag(xb + (size_t)row0 * DD + dk, DD, lane);
        v16bf b = load_b_frag(wT + (size_t)col0 * DD + dk, DD, lane);
        c = wmma_bf16(a, b, c);
    }

    const int n = lane & 15, half = lane >> 4;
    const int col = col0 + n;
    const float bc = bias[col];
    if (!transpose_out) {
#pragma unroll
        for (int r = 0; r < 8; ++r) {
            int row = row0 + r + half * 8;
            out_rm[(size_t)row * DD + col] = (bf16_t)(c[r] + bc);
        }
    } else {
#pragma unroll
        for (int r = 0; r < 8; ++r) {
            int row = row0 + r + half * 8;      // global row over B*N
            int bidx = row >> 11;               // / NN
            int nn = row & (NN - 1);
            out_tr[(size_t)bidx * DD * NN + (size_t)col * NN + nn] =
                (bf16_t)(c[r] + bc);
        }
    }
}

// ---------------------------------------------------------------------------
// Flash attention + fused output projection.
// One wave per (batch, 16-query tile). Context (16x256 f32) lives in 128
// accumulator VGPRs; softmax state replicated per 16-lane half; never
// materializes the NxN score matrix.
// ---------------------------------------------------------------------------
__global__ __launch_bounds__(32)
void attn_kernel(const bf16_t* __restrict__ Q, const bf16_t* __restrict__ K,
                 const bf16_t* __restrict__ VT, const float* __restrict__ Ww,
                 const float* __restrict__ bw, float* __restrict__ out) {
    __shared__ bf16_t q_lds[16 * DD];   // 8 KB: this wave's Q tile
    __shared__ bf16_t p_lds[16 * 32];   // 1 KB: P relayout C->A fragment

    const int lane = threadIdx.x;
    const int tiles_per_b = NN / 16;
    const int b  = blockIdx.x / tiles_per_b;
    const int qt = blockIdx.x % tiles_per_b;
    const size_t qrow0 = (size_t)b * NN + (size_t)qt * 16;

    // Stage Q tile (16 x 256 bf16) into LDS
    for (int i = lane; i < 16 * DD; i += 32)
        q_lds[i] = Q[(qrow0 + (i >> 8)) * DD + (i & (DD - 1))];
    __syncthreads();

    const bf16_t* Kpb  = K  + (size_t)b * NN * DD;  // [N][D] row-major
    const bf16_t* VTpb = VT + (size_t)b * DD * NN;  // [D][N]

    v8f o[16];
#pragma unroll
    for (int t = 0; t < 16; ++t) { v8f z = {}; o[t] = z; }
    float mstate[8], lstate[8];
#pragma unroll
    for (int r = 0; r < 8; ++r) { mstate[r] = -1e30f; lstate[r] = 0.f; }

    const int half = lane >> 4;
    const float scale = 0.0625f;        // 1/sqrt(256)

#pragma unroll 1
    for (int n0 = 0; n0 < NN; n0 += 32) {
        // ---- S = Q K^T for 32 keys (two 16x16 C tiles) ----
        v8f s0 = {}, s1 = {};
#pragma unroll
        for (int dk = 0; dk < DD; dk += 32) {
            v16bf a   = load_a_frag(q_lds + dk, DD, lane);
            v16bf bk0 = load_b_frag(Kpb + (size_t)n0 * DD + dk, DD, lane);
            v16bf bk1 = load_b_frag(Kpb + (size_t)(n0 + 16) * DD + dk, DD, lane);
            s0 = wmma_bf16(a, bk0, s0);
            s1 = wmma_bf16(a, bk1, s1);
        }

        // ---- online softmax (row = M = r + half*8; reductions stay in half) ----
        float alpha[8];
#pragma unroll
        for (int r = 0; r < 8; ++r) {
            float v0 = s0[r] * scale;
            float v1 = s1[r] * scale;
            float tm = fmaxf(v0, v1);
#pragma unroll
            for (int m = 8; m >= 1; m >>= 1) tm = fmaxf(tm, __shfl_xor(tm, m, 32));
            float mnew = fmaxf(mstate[r], tm);
            alpha[r] = __expf(mstate[r] - mnew);
            float p0 = __expf(v0 - mnew);
            float p1 = __expf(v1 - mnew);
            float rs = p0 + p1;
#pragma unroll
            for (int m = 8; m >= 1; m >>= 1) rs += __shfl_xor(rs, m, 32);
            lstate[r] = lstate[r] * alpha[r] + rs;
            mstate[r] = mnew;
            int M = r + half * 8;
            p_lds[M * 32 + (lane & 15)]      = (bf16_t)p0;
            p_lds[M * 32 + 16 + (lane & 15)] = (bf16_t)p1;
        }
        __syncthreads();

        v16bf pf = load_a_frag(p_lds, 32, lane);   // P as 16x32 A-fragment

        // ---- O = diag(alpha)*O + P * V  (16 column tiles of D) ----
#pragma unroll
        for (int t = 0; t < 16; ++t) {
#pragma unroll
            for (int r = 0; r < 8; ++r) o[t][r] *= alpha[r];
            v16bf bv = load_b_frag(VTpb + (size_t)(t * 16) * NN + n0, NN, lane);
            o[t] = wmma_bf16(pf, bv, o[t]);
        }
        __syncthreads();   // protect p_lds for next iteration
    }

    // ---- fused final projection: out = (O/l) . Ww + bw ----
    float acc[8] = {0, 0, 0, 0, 0, 0, 0, 0};
#pragma unroll
    for (int t = 0; t < 16; ++t) {
        float w = Ww[t * 16 + (lane & 15)];
#pragma unroll
        for (int r = 0; r < 8; ++r) acc[r] += o[t][r] * w;
    }
#pragma unroll
    for (int r = 0; r < 8; ++r) {
        float a = acc[r] / lstate[r];
#pragma unroll
        for (int m = 8; m >= 1; m >>= 1) a += __shfl_xor(a, m, 32);
        if ((lane & 15) == 0)
            out[qrow0 + r + half * 8] = a + bw[0];
    }
}

// ---------------------------------------------------------------------------
// Launch
// ---------------------------------------------------------------------------
extern "C" void kernel_launch(void* const* d_in, const int* in_sizes, int n_in,
                              void* d_out, int out_size, void* d_ws, size_t ws_size,
                              hipStream_t stream) {
    const float* x  = (const float*)d_in[0];
    const float* Wq = (const float*)d_in[1];
    const float* bq = (const float*)d_in[2];
    const float* Wk = (const float*)d_in[3];
    const float* bk = (const float*)d_in[4];
    const float* Wv = (const float*)d_in[5];
    const float* bv = (const float*)d_in[6];
    const float* Ww = (const float*)d_in[7];
    const float* bw = (const float*)d_in[8];
    float* out = (float*)d_out;

    // Workspace carve-up (~135 MB total)
    char* ws = (char*)d_ws;
    const size_t nx = (size_t)BB * NN * DD;          // 16,777,216 elems
    bf16_t* xb  = (bf16_t*)ws; ws += nx * sizeof(bf16_t);
    bf16_t* Qb  = (bf16_t*)ws; ws += nx * sizeof(bf16_t);
    bf16_t* Kb  = (bf16_t*)ws; ws += nx * sizeof(bf16_t);
    bf16_t* VTb = (bf16_t*)ws; ws += nx * sizeof(bf16_t);
    bf16_t* WqT = (bf16_t*)ws; ws += (size_t)DD * DD * sizeof(bf16_t);
    bf16_t* WkT = (bf16_t*)ws; ws += (size_t)DD * DD * sizeof(bf16_t);
    bf16_t* WvT = (bf16_t*)ws; ws += (size_t)DD * DD * sizeof(bf16_t);

    // 1) precision conversion (x -> bf16, W -> bf16 transposed)
    cvt_bf16_kernel<<<(int)(nx / 256), 256, 0, stream>>>(x, xb, (int)nx);
    cvt_wT_kernel<<<(DD * DD) / 256, 256, 0, stream>>>(Wq, WqT);
    cvt_wT_kernel<<<(DD * DD) / 256, 256, 0, stream>>>(Wk, WkT);
    cvt_wT_kernel<<<(DD * DD) / 256, 256, 0, stream>>>(Wv, WvT);

    // 2) Q, K (row-major) and V^T projections via WMMA
    dim3 gproj(BB * NN / 16, DD / 16);
    qkv_proj_kernel<<<gproj, 32, 0, stream>>>(xb, WqT, bq, Qb, nullptr, 0);
    qkv_proj_kernel<<<gproj, 32, 0, stream>>>(xb, WkT, bk, Kb, nullptr, 0);
    qkv_proj_kernel<<<gproj, 32, 0, stream>>>(xb, WvT, bv, nullptr, VTb, 1);

    // 3) flash attention + fused output projection
    attn_kernel<<<BB * NN / 16, 32, 0, stream>>>(Qb, Kb, VTb, Ww, bw, out);
}
